// SAGEConvLayer_70935679861287
// MI455X (gfx1250) — compile-verified, matrix operand order checked by reading, not run on hardware
//
#include <hip/hip_runtime.h>
#include <math.h>

// ---------------------------------------------------------------------------
// SAGEConv layer for MI455X (gfx1250, wave32, WMMA 16x16x32 f16->f32)
//   h    = relu(x @ Wp^T + bp)                 (h stored f16 for the gather)
//   agg  = segment_sum(h[src], dst) (f32 atomics);  cnt = segment_sum(1, dst)
//   mean = agg / max(cnt, 1)
//   out  = elu(mean @ Wl^T + bl + x @ Wr^T)
//   out  = x + out / max(||out||_2, 1e-12)
// Weights pre-converted to f16 once so GEMM B-fragments load directly as
// 16-byte f16 chunks (no per-block f32->f16 conversion of the weights).
// ---------------------------------------------------------------------------

typedef __attribute__((ext_vector_type(16))) _Float16 v16h;
typedef __attribute__((ext_vector_type(4)))  _Float16 v4h;
typedef __attribute__((ext_vector_type(8)))  float    v8f;

#define DIMS 128
#define EPS  1e-12f

// ---- WMMA fragment loaders (wave32 layouts per CDNA5 ISA 7.12.2) ----------
// A-matrix 16x32 f16: lane L holds row M = L&15; VGPRs hold K with the upper
// half-wave offset by 8:  e in [0,8)  -> K = hs + e
//                         e in [8,16) -> K = 16 + hs + (e-8)
// where hs = (L>>4)*8.  B-matrix 32x16 mirrors this with lane = column N.

__device__ inline v16h load_frag_f32(const float* __restrict__ base, int ld,
                                     int k0, int lane) {
  const int m  = lane & 15;
  const int hs = (lane >> 4) * 8;
  const float* row = base + (size_t)m * ld + k0;
  v16h a;
#pragma unroll
  for (int e = 0; e < 8; ++e) a[e]     = (_Float16)row[hs + e];
#pragma unroll
  for (int e = 0; e < 8; ++e) a[8 + e] = (_Float16)row[16 + hs + e];
  return a;
}

__device__ inline v16h load_frag_f32_scaled(const float* __restrict__ base,
                                            int ld, int k0, int lane,
                                            float scale) {
  const int m  = lane & 15;
  const int hs = (lane >> 4) * 8;
  const float* row = base + (size_t)m * ld + k0;
  v16h a;
#pragma unroll
  for (int e = 0; e < 8; ++e) a[e]     = (_Float16)(row[hs + e] * scale);
#pragma unroll
  for (int e = 0; e < 8; ++e) a[8 + e] = (_Float16)(row[16 + hs + e] * scale);
  return a;
}

// direct f16 fragment: two contiguous 16-byte chunks per lane -> b128 loads
__device__ inline v16h load_frag_f16(const _Float16* __restrict__ base, int ld,
                                     int k0, int lane) {
  const int m  = lane & 15;
  const int hs = (lane >> 4) * 8;
  const _Float16* row = base + (size_t)m * ld + k0;
  v16h a;
#pragma unroll
  for (int e = 0; e < 8; ++e) a[e]     = row[hs + e];
#pragma unroll
  for (int e = 0; e < 8; ++e) a[8 + e] = row[16 + hs + e];
  return a;
}

// ---- kernel 0: clear agg/cnt + convert the 3 weight matrices to f16 -------
__global__ void sage_prep_kernel(const float* __restrict__ Wp,
                                 const float* __restrict__ Wl,
                                 const float* __restrict__ Wr,
                                 _Float16* __restrict__ w16,   // [3*D*D]
                                 float* __restrict__ agg,
                                 float* __restrict__ cnt,
                                 int n_agg, int n_cnt) {
  const int stride = gridDim.x * blockDim.x;
  const int tid = blockIdx.x * blockDim.x + threadIdx.x;
  for (int i = tid; i < DIMS * DIMS; i += stride) {
    w16[i]                   = (_Float16)Wp[i];
    w16[DIMS * DIMS + i]     = (_Float16)Wl[i];
    w16[2 * DIMS * DIMS + i] = (_Float16)Wr[i];
  }
  for (int i = tid; i < n_agg; i += stride) agg[i] = 0.0f;
  for (int i = tid; i < n_cnt; i += stride) cnt[i] = 0.0f;
}

// ---- kernel 1: h16 = relu(x @ Wp^T + bp) ----------------------------------
// block = 256 threads = 8 waves; block owns 16 rows, wave w owns cols [16w,16w+16)
__global__ void __launch_bounds__(256)
sage_project_kernel(const float* __restrict__ x,
                    const _Float16* __restrict__ Wp16,
                    const float* __restrict__ bp,
                    _Float16* __restrict__ h16) {
  const int row0 = blockIdx.x * 16;
  const int wave = threadIdx.x >> 5;
  const int lane = threadIdx.x & 31;
  const int col0 = wave * 16;

  const float*    abase = x    + (size_t)row0 * DIMS;
  const _Float16* bbase = Wp16 + (size_t)col0 * DIMS;  // row j of Wp = col j of Wp^T

  v8f c = {};
#pragma unroll
  for (int kt = 0; kt < DIMS; kt += 32) {
    v16h a = load_frag_f32(abase, DIMS, kt, lane);
    v16h b = load_frag_f16(bbase, DIMS, kt, lane);
    c = __builtin_amdgcn_wmma_f32_16x16x32_f16(false, a, false, b,
                                               (short)0, c, false, false);
  }

  const int n    = lane & 15;
  const int mrow = (lane >> 4) * 8;
  const float bias = bp[col0 + n];
#pragma unroll
  for (int r = 0; r < 8; ++r) {
    float v = c[r] + bias;
    v = v > 0.0f ? v : 0.0f;                           // relu
    h16[(size_t)(row0 + mrow + r) * DIMS + col0 + n] = (_Float16)v;
  }
}

// ---- kernel 2: scatter mean-aggregation -----------------------------------
// one wave per edge: 32 lanes x 4 dims; f16 gather (b64/lane), f32 atomics.
// h16 (12.8 MB) and agg (25.6 MB) are L2-resident (192 MB L2), so the random
// gather hits L2 and the f32 atomics resolve in L2's atomic units.
__global__ void __launch_bounds__(256)
sage_scatter_kernel(const _Float16* __restrict__ h16,
                    const int* __restrict__ src,
                    const int* __restrict__ dst,
                    float* __restrict__ agg,
                    float* __restrict__ cnt,
                    int n_edges) {
  const int e = blockIdx.x * 8 + (threadIdx.x >> 5);
  if (e >= n_edges) return;
  const int lane = threadIdx.x & 31;
  const int s = src[e];
  const int d = dst[e];
  const v4h v = *(const v4h*)(h16 + (size_t)s * DIMS + lane * 4);
  float* a = agg + (size_t)d * DIMS + lane * 4;
  atomicAdd(a + 0, (float)v[0]);
  atomicAdd(a + 1, (float)v[1]);
  atomicAdd(a + 2, (float)v[2]);
  atomicAdd(a + 3, (float)v[3]);
  if (lane == 0) atomicAdd(cnt + d, 1.0f);
}

// ---- kernel 3: combine + ELU + L2-normalize + residual --------------------
// block owns 16 rows x 128 cols; dual GEMM accumulated into one WMMA acc;
// post-GEMM tile staged in LDS for the row-wise L2 norm.
__global__ void __launch_bounds__(256)
sage_combine_kernel(const float* __restrict__ x,
                    const float* __restrict__ agg,
                    const float* __restrict__ cnt,
                    const _Float16* __restrict__ Wl16,
                    const float* __restrict__ bl,
                    const _Float16* __restrict__ Wr16,
                    float* __restrict__ out) {
  __shared__ float tile[16][DIMS];
  __shared__ float partial[16][16];
  __shared__ float inv_norm[16];

  const int row0 = blockIdx.x * 16;
  const int wave = threadIdx.x >> 5;
  const int lane = threadIdx.x & 31;
  const int col0 = wave * 16;
  const int m    = lane & 15;

  // per-lane mean scale for the row this lane supplies to the A fragment
  const float inv_cnt = 1.0f / fmaxf(cnt[row0 + m], 1.0f);

  const float*    agg_base = agg  + (size_t)row0 * DIMS;
  const float*    x_base   = x    + (size_t)row0 * DIMS;
  const _Float16* wl_base  = Wl16 + (size_t)col0 * DIMS;
  const _Float16* wr_base  = Wr16 + (size_t)col0 * DIMS;

  v8f c = {};
#pragma unroll
  for (int kt = 0; kt < DIMS; kt += 32) {              // mean @ Wl^T
    v16h a = load_frag_f32_scaled(agg_base, DIMS, kt, lane, inv_cnt);
    v16h b = load_frag_f16(wl_base, DIMS, kt, lane);
    c = __builtin_amdgcn_wmma_f32_16x16x32_f16(false, a, false, b,
                                               (short)0, c, false, false);
  }
#pragma unroll
  for (int kt = 0; kt < DIMS; kt += 32) {              // + x @ Wr^T
    v16h a = load_frag_f32(x_base, DIMS, kt, lane);
    v16h b = load_frag_f16(wr_base, DIMS, kt, lane);
    c = __builtin_amdgcn_wmma_f32_16x16x32_f16(false, a, false, b,
                                               (short)0, c, false, false);
  }

  const int n    = lane & 15;
  const int mrow = (lane >> 4) * 8;
  const float bias = bl[col0 + n];
#pragma unroll
  for (int r = 0; r < 8; ++r) {
    float v = c[r] + bias;
    v = v > 0.0f ? v : (__expf(v) - 1.0f);             // ELU (alpha = 1)
    tile[mrow + r][col0 + n] = v;
  }
  __syncthreads();

  // row-wise sum of squares: 16 threads per row, 8 elements each
  {
    const int r   = threadIdx.x >> 4;     // 0..15
    const int seg = threadIdx.x & 15;     // 0..15
    float s = 0.0f;
#pragma unroll
    for (int j = 0; j < 8; ++j) {
      const float v = tile[r][seg * 8 + j];
      s += v * v;
    }
    partial[r][seg] = s;
  }
  __syncthreads();

  if (threadIdx.x < 16) {
    float s = 0.0f;
#pragma unroll
    for (int j = 0; j < 16; ++j) s += partial[threadIdx.x][j];
    inv_norm[threadIdx.x] = 1.0f / fmaxf(sqrtf(s), EPS);
  }
  __syncthreads();

  // residual write: 2048 elements / 256 threads = 8 each
  for (int i = threadIdx.x; i < 16 * DIMS; i += 256) {
    const int r = i >> 7;
    const int q = i & (DIMS - 1);
    const size_t idx = (size_t)(row0 + r) * DIMS + q;
    out[idx] = x[idx] + tile[r][q] * inv_norm[r];
  }
}

// ---------------------------------------------------------------------------
extern "C" void kernel_launch(void* const* d_in, const int* in_sizes, int n_in,
                              void* d_out, int out_size, void* d_ws, size_t ws_size,
                              hipStream_t stream) {
  const float* x  = (const float*)d_in[0];
  const int*   ei = (const int*)d_in[1];          // [2, E] int32 (JAX x64 off)
  const float* Wp = (const float*)d_in[2];
  const float* bp = (const float*)d_in[3];
  const float* Wl = (const float*)d_in[4];
  const float* bl = (const float*)d_in[5];
  const float* Wr = (const float*)d_in[6];
  float* out = (float*)d_out;

  const int N = in_sizes[0] / DIMS;               // 50000
  const int E = in_sizes[1] / 2;                  // 800000
  const int* src = ei;
  const int* dst = ei + E;

  // workspace layout (f32-aligned):
  //   agg[N*D] f32 | cnt[N] f32 | W16[3*D*D] f16 | h16[N*D] f16   (~38.7 MB)
  float*    agg  = (float*)d_ws;
  float*    cnt  = agg + (size_t)N * DIMS;
  _Float16* w16  = (_Float16*)(cnt + N);
  _Float16* wp16 = w16;
  _Float16* wl16 = w16 + DIMS * DIMS;
  _Float16* wr16 = w16 + 2 * DIMS * DIMS;
  _Float16* h16  = w16 + 3 * DIMS * DIMS;

  const int row_tiles = (N + 15) / 16;            // N = 3125 * 16 exactly

  sage_prep_kernel<<<2048, 256, 0, stream>>>(Wp, Wl, Wr, w16, agg, cnt,
                                             N * DIMS, N);

  sage_project_kernel<<<row_tiles, 256, 0, stream>>>(x, wp16, bp, h16);

  sage_scatter_kernel<<<(E + 7) / 8, 256, 0, stream>>>(h16, src, dst, agg, cnt, E);

  sage_combine_kernel<<<row_tiles, 256, 0, stream>>>(x, agg, cnt, wl16, bl,
                                                     wr16, out);
}
